// bi_convlstm_based_att_with_refinement_21801253994831
// MI455X (gfx1250) — matrix operand order repaired and checked by reading, not compiled, not run on hardware
//
#include <hip/hip_runtime.h>
#include <math.h>

// CDNA5 / gfx1250: wave32, WMMA 16x16x32 f16->f32, all LDS staging via
// ASYNCcnt async global->LDS gathers, vectorized (b128) fragment reads.
typedef __attribute__((ext_vector_type(16))) _Float16 v16h;
typedef __attribute__((ext_vector_type(8)))  float    v8f;
typedef __attribute__((ext_vector_type(4)))  float    v4f;

#define WAVES_PER_BLOCK 8

// Fixed workload shape (B=16, H=W=256).
#define IMG_H 256
#define IMG_W 256
#define BATCH 16

__device__ __forceinline__ float sigmf(float x) { return 1.0f / (1.0f + __expf(-x)); }
__device__ __forceinline__ float tanhf_fast(float x) {
    const float e = __expf(2.0f * x);
    return (e - 1.0f) / (e + 1.0f);
}

// Implicit-GEMM convolution, fused epilogue.
//   EPI = 0 : ConvLSTM gates (COUT == 32 = [i|f|o|g] x 8)
//   EPI = 1 : ReLU      EPI = 2 : PReLU      EPI = 3 : sigmoid
// Input = channel-concat of in0 (CIN0 ch) and in1 (CIN1 ch); null in1 == zeros.
// zpad: small zeroed global region; every OOB/padding gather is redirected there
// so each im2col element issues exactly one async load (exact ASYNCcnt bookkeeping).
template<int KS, int EPI, int COUT, int CIN0, int CIN1>
__global__ __launch_bounds__(32 * WAVES_PER_BLOCK)
void conv_wmma_kernel(const float* __restrict__ in0, long in0bs,
                      const float* __restrict__ in1, long in1bs,
                      const float* __restrict__ wgt, const float* __restrict__ bias,
                      const float* __restrict__ cprev,
                      float* __restrict__ outh, long outbs, int outcoff,
                      float* __restrict__ outc,
                      const float* __restrict__ prelu,
                      const float* __restrict__ zpad)
{
    constexpr int PAD  = KS / 2;
    constexpr int CIN  = CIN0 + CIN1;
    constexpr int K    = CIN * KS * KS;
    constexpr int NCH  = (K + 31) / 32;             // 32-wide K chunks
    constexpr int KPAD = NCH * 32;
    constexpr int NB   = (COUT > 16) ? 32 : 16;     // weight/Z tile width
    constexpr long HW  = (long)IMG_H * IMG_W;

    const int lane = threadIdx.x & 31;
    const int wid  = threadIdx.x >> 5;
    // nTiles = BATCH*256*16 = 65536, exact multiple of WAVES_PER_BLOCK.
    const int tile = blockIdx.x * WAVES_PER_BLOCK + wid;
    const int tx = tile & 15;                        // 16 tiles per row
    const int y  = (tile >> 4) & (IMG_H - 1);
    const int bb = tile >> 12;
    const int x0 = tx << 4;

    __shared__ float Ash[WAVES_PER_BLOCK][2][16][32];  // double-buffered im2col (f32)
    __shared__ float Bsh[NB][KPAD];                    // weights TRANSPOSED [n][k] (block-shared)
    __shared__ float Zsh[WAVES_PER_BLOCK][16][NB];     // epilogue staging [pixel][n]

    // ---- stage ALL weights once per block, async global->LDS ----
    for (int i = threadIdx.x; i < NB * KPAD; i += 32 * WAVES_PER_BLOCK) {
        const int n = i / KPAD;
        const int k = i - n * KPAD;
        const float* src = (n < COUT && k < K) ? (wgt + (long)n * K + k) : zpad;
        const unsigned laddr = (unsigned)(size_t)&Bsh[n][k];
        asm volatile("global_load_async_to_lds_b32 %0, %1, off"
                     :: "v"(laddr), "v"(src) : "memory");
    }
    asm volatile("s_wait_asynccnt 0x0" ::: "memory");  // own loads done; barrier covers peers
    __syncthreads();

    // ---- async im2col stage: 16 ops/wave/chunk, k decoded once per lane ----
    auto stageA = [&](int kc) {
        const int buf = kc & 1;
        const int k   = (kc << 5) + lane;
        const float* base = nullptr;
        int dx = 0;
        if (k < K) {
            const int c  = k / (KS * KS);
            const int r  = k - c * (KS * KS);
            const int ry = r / KS;
            dx = (r - ry * KS) - PAD;
            const int yy = y + ry - PAD;
            if (yy >= 0 && yy < IMG_H) {
                if (c < CIN0)
                    base = in0 + (long)bb * in0bs + (long)c * HW + (long)yy * IMG_W;
                else if (in1)
                    base = in1 + (long)bb * in1bs + (long)(c - CIN0) * HW + (long)yy * IMG_W;
            }
        }
        #pragma unroll
        for (int m = 0; m < 16; ++m) {
            const int xx = x0 + m + dx;
            const float* src = (base && xx >= 0 && xx < IMG_W) ? (base + xx) : zpad;
            const unsigned laddr = (unsigned)(size_t)&Ash[wid][buf][m][lane];
            // gfx1250 async VMEM: lane-scattered global gather landing in LDS.
            asm volatile("global_load_async_to_lds_b32 %0, %1, off"
                         :: "v"(laddr), "v"(src) : "memory");
        }
    };

    v8f acc0 = {};
    v8f acc1 = {};

    stageA(0);
    for (int kc = 0; kc < NCH; ++kc) {
        if (kc + 1 < NCH) {
            stageA(kc + 1);
            // async loads retire in order: <=16 outstanding => chunk kc landed,
            // chunk kc+1 (16 newest) still in flight.
            asm volatile("s_wait_asynccnt 0x10" ::: "memory");
        } else {
            asm volatile("s_wait_asynccnt 0x0" ::: "memory");
        }
        __builtin_amdgcn_wave_barrier();

        // ---- fragments in contiguous k order (same for A and B), b128 LDS reads ----
        // A 16x32: lane holds M=lane&15; K = kbase+e, kbase=(lane&16)
        //   -> 16 contiguous floats: [kbase..kbase+15]
        // B 32x16: lane holds N=lane&15; K = (lane&16)+e -> 16 contiguous floats of Bsh[n][.]
        const int buf = kc & 1;
        const int k0  = kc << 5;
        const int mA  = lane & 15;
        const int kbA = (lane & 16);
        const int kbB = (lane & 16);
        const int nB  = lane & 15;

        const v4f* arow = (const v4f*)&Ash[wid][buf][mA][0];
        const int  q    = kbA >> 2;                   // 0 or 4
        const v4f a0q = arow[q + 0];
        const v4f a1q = arow[q + 1];
        const v4f a2q = arow[q + 2];
        const v4f a3q = arow[q + 3];

        const v4f* brow0 = (const v4f*)&Bsh[nB][k0 + kbB];
        const v4f b0q0 = brow0[0], b0q1 = brow0[1], b0q2 = brow0[2], b0q3 = brow0[3];

        v16h af = {}, bf0 = {}, bf1 = {};
        #pragma unroll
        for (int e = 0; e < 4; ++e) {
            af[e]      = (_Float16)a0q[e];
            af[4 + e]  = (_Float16)a1q[e];
            af[8 + e]  = (_Float16)a2q[e];
            af[12 + e] = (_Float16)a3q[e];
            bf0[e]      = (_Float16)b0q0[e];
            bf0[4 + e]  = (_Float16)b0q1[e];
            bf0[8 + e]  = (_Float16)b0q2[e];
            bf0[12 + e] = (_Float16)b0q3[e];
        }
        if constexpr (COUT > 16) {
            const v4f* brow1 = (const v4f*)&Bsh[16 + nB][k0 + kbB];
            const v4f b1q0 = brow1[0], b1q1 = brow1[1], b1q2 = brow1[2], b1q3 = brow1[3];
            #pragma unroll
            for (int e = 0; e < 4; ++e) {
                bf1[e]      = (_Float16)b1q0[e];
                bf1[4 + e]  = (_Float16)b1q1[e];
                bf1[8 + e]  = (_Float16)b1q2[e];
                bf1[12 + e] = (_Float16)b1q3[e];
            }
        }

        acc0 = __builtin_amdgcn_wmma_f32_16x16x32_f16(
                   false, af, false, bf0, (short)0, acc0, false, false);
        if constexpr (COUT > 16) {
            acc1 = __builtin_amdgcn_wmma_f32_16x16x32_f16(
                       false, af, false, bf1, (short)0, acc1, false, false);
        }
        __builtin_amdgcn_wave_barrier();
    }

    // ---- spill accumulators (D layout) to Zsh[pixel][channel], add bias ----
    {
        const int nD = lane & 15;
        const int m0 = (lane >> 4) << 3;
        const float b0 = (nD < COUT) ? bias[nD] : 0.0f;
        const float b1 = (COUT > 16) ? bias[16 + nD] : 0.0f;
        #pragma unroll
        for (int r = 0; r < 8; ++r) {
            Zsh[wid][m0 + r][nD] = acc0[r] + b0;
            if constexpr (COUT > 16) Zsh[wid][m0 + r][16 + nD] = acc1[r] + b1;
        }
    }
    __builtin_amdgcn_wave_barrier();

    // ---- fused epilogue ----
    if constexpr (EPI == 0) {
        // ConvLSTM: z = [i(8) | f(8) | o(8) | g(8)]
        #pragma unroll
        for (int i = lane; i < 16 * 8; i += 32) {
            const int m = i >> 3;
            const int h = i & 7;
            const float zi = Zsh[wid][m][h];
            const float zf = Zsh[wid][m][8 + h];
            const float zo = Zsh[wid][m][16 + h];
            const float zg = Zsh[wid][m][24 + h];
            const long cidx = ((long)bb * 8 + h) * HW + (long)y * IMG_W + (x0 + m);
            const float cp = cprev ? cprev[cidx] : 0.0f;
            const float cn = sigmf(zf) * cp + sigmf(zi) * tanhf_fast(zg);
            const float hn = sigmf(zo) * tanhf_fast(cn);
            outc[cidx] = cn;
            outh[(long)bb * outbs + (long)(outcoff + h) * HW + (long)y * IMG_W + (x0 + m)] = hn;
        }
    } else {
        const float aslope = (EPI == 2) ? prelu[0] : 0.0f;
        #pragma unroll
        for (int i = lane; i < 16 * COUT; i += 32) {
            const int m = i / COUT;
            const int n = i - m * COUT;
            float v = Zsh[wid][m][n];
            if constexpr (EPI == 1)      v = fmaxf(v, 0.0f);
            else if constexpr (EPI == 2) v = (v >= 0.0f) ? v : aslope * v;
            else if constexpr (EPI == 3) v = sigmf(v);
            outh[(long)bb * outbs + (long)(outcoff + n) * HW + (long)y * IMG_W + (x0 + m)] = v;
        }
    }
}

template<int KS, int EPI, int COUT, int CIN0, int CIN1>
static void conv_launch(hipStream_t st,
                        const float* in0, long in0bs,
                        const float* in1, long in1bs,
                        const float* w, const float* b,
                        const float* cp, float* oh, long obs, int coff, float* oc,
                        const float* prelu, const float* zpad)
{
    const int nTiles = BATCH * IMG_H * (IMG_W / 16);   // 65536
    const int blocks = nTiles / WAVES_PER_BLOCK;
    conv_wmma_kernel<KS, EPI, COUT, CIN0, CIN1><<<blocks, 32 * WAVES_PER_BLOCK, 0, st>>>(
        in0, in0bs, in1, in1bs, w, b, cp, oh, obs, coff, oc, prelu, zpad);
}

extern "C" void kernel_launch(void* const* d_in, const int* in_sizes, int n_in,
                              void* d_out, int out_size, void* d_ws, size_t ws_size,
                              hipStream_t stream)
{
    (void)in_sizes; (void)n_in; (void)out_size; (void)ws_size;
    const int B = BATCH, T = 7;
    const long HW = (long)IMG_H * IMG_W;

    const float* x     = (const float*)d_in[0];   // [B,7,4,H,W]
    const float* recon = (const float*)d_in[1];   // [B,1,H,W]
    const float* lw[5] = {(const float*)d_in[2], (const float*)d_in[4],
                          (const float*)d_in[6], (const float*)d_in[8],
                          (const float*)d_in[10]};
    const float* lb[5] = {(const float*)d_in[3], (const float*)d_in[5],
                          (const float*)d_in[7], (const float*)d_in[9],
                          (const float*)d_in[11]};
    const float* c0w = (const float*)d_in[12]; const float* c0b = (const float*)d_in[13];
    const float* c1w = (const float*)d_in[14]; const float* c1b = (const float*)d_in[15];
    const float* c2w = (const float*)d_in[16]; const float* c2b = (const float*)d_in[17];
    const float* r0w = (const float*)d_in[18]; const float* r0b = (const float*)d_in[19];
    const float* r1w = (const float*)d_in[20]; const float* r1b = (const float*)d_in[21];
    const float* pa0 = (const float*)d_in[22]; const float* pa1 = (const float*)d_in[23];
    float* out = (float*)d_out;                   // [B,1,H,W]

    // Workspace layout (floats):
    //   zpad  : 64 zeroed floats (async-gather padding target)
    //   stfea : 16*B*HW   (fwd hidden ch 0..7, bwd hidden ch 8..15)
    //   pool  : 7 * 8*B*HW  -> seq ping-pong A(3)/B(3) + cbuf; reused post-LSTM
    float* ws    = (float*)d_ws;
    float* zpad  = ws;
    float* stfea = ws + 64;
    float* pool  = stfea + (long)16 * B * HW;
    float* seq[6];
    for (int i = 0; i < 6; ++i) seq[i] = pool + (long)i * 8 * B * HW;
    float* cbuf = pool + (long)6 * 8 * B * HW;

    hipMemsetAsync(zpad, 0, 64 * sizeof(float), stream);   // capture-legal

    // ---- bidirectional 5-layer ConvLSTM scan (3 steps each direction) ----
    for (int dir = 0; dir < 2; ++dir) {
        for (int L = 0; L < 5; ++L) {
            const int outBase = (L & 1) ? 3 : 0;   // even layers write set A
            const int inBase  = (L & 1) ? 0 : 3;   // ... read the other set
            for (int t = 0; t < 3; ++t) {
                const float* h_prev = (t == 0) ? nullptr : seq[outBase + (t - 1)];
                const float* cp     = (t == 0) ? nullptr : cbuf;
                float* oh; long obs; int coff;
                if (L == 4 && t == 2) { oh = stfea;            obs = 16 * HW; coff = dir * 8; }
                else                  { oh = seq[outBase + t]; obs = 8 * HW;  coff = 0; }
                if (L == 0) {
                    const int tsel = (dir == 0) ? t : (6 - t);
                    conv_launch<3, 0, 32, 4, 8>(stream,
                        x + (long)tsel * 4 * HW, (long)T * 4 * HW, h_prev, 8 * HW,
                        lw[0], lb[0], cp, oh, obs, coff, cbuf, nullptr, zpad);
                } else {
                    conv_launch<3, 0, 32, 8, 8>(stream,
                        seq[inBase + t], 8 * HW, h_prev, 8 * HW,
                        lw[L], lb[L], cp, oh, obs, coff, cbuf, nullptr, zpad);
                }
            }
        }
    }

    // ---- refinement branch + attention head (pool is free now) ----
    float* r1 = pool;
    float* r2 = r1 + (long)16 * B * HW;
    float* a0 = r2 + (long)16 * B * HW;
    float* a1 = a0 + (long)8  * B * HW;

    conv_launch<3, 1, 16, 1, 0>(stream, recon, HW, nullptr, 0, r0w, r0b,
                                nullptr, r1, 16 * HW, 0, nullptr, nullptr, zpad);
    conv_launch<3, 1, 16, 16, 0>(stream, r1, 16 * HW, nullptr, 0, r1w, r1b,
                                 nullptr, r2, 16 * HW, 0, nullptr, nullptr, zpad);
    // a = cat([st_fea(16), r(16)]) -> conv0 5x5 -> PReLU
    conv_launch<5, 2, 8, 16, 16>(stream, stfea, 16 * HW, r2, 16 * HW, c0w, c0b,
                                 nullptr, a0, 8 * HW, 0, nullptr, pa0, zpad);
    conv_launch<5, 2, 4, 8, 0>(stream, a0, 8 * HW, nullptr, 0, c1w, c1b,
                               nullptr, a1, 4 * HW, 0, nullptr, pa1, zpad);
    conv_launch<5, 3, 1, 4, 0>(stream, a1, 4 * HW, nullptr, 0, c2w, c2b,
                               nullptr, out, 1 * HW, 0, nullptr, nullptr, zpad);
}
